// Backbone_66546223284373
// MI455X (gfx1250) — matrix operand order repaired and checked by reading, not compiled, not run on hardware
//
#include <hip/hip_runtime.h>
#include <hip/hip_bf16.h>
#include <stdint.h>

// ---------------------------------------------------------------------------
// GCN backbone on MI455X (gfx1250): f16 WMMA (f32 accumulate) for all GEMMs,
// weights staged into LDS with the Tensor Data Mover (double-buffered, one
// 32KB contiguous stage per 128x128 GEMM slice), activations resident in LDS.
// One workgroup = 16 graphs = 144 rows = 9 WMMA M-tiles. 256 threads = 8 waves.
// ---------------------------------------------------------------------------

typedef __attribute__((ext_vector_type(16))) _Float16 v16h;
typedef __attribute__((ext_vector_type(2)))  _Float16 h2;
typedef __attribute__((ext_vector_type(8)))  float    v8f;
typedef __attribute__((ext_vector_type(4)))  unsigned int u32x4;
typedef __attribute__((ext_vector_type(8)))  int      i32x8;
typedef __attribute__((ext_vector_type(4)))  int      i32x4;

#define BB      32768
#define NN      9
#define RR      128
#define FF      512
#define GPB     16            // graphs per block
#define ROWS    144           // GPB * NN
#define MT      9             // ROWS / 16
#define LDA     136           // halves, padded row stride for f16 buffers
#define LDC     132           // floats, padded row stride for f32 buffer
#define NWAVES  8

// swizzled weight tile bases (units of 512-half = 1KB tiles)
#define GCN_BASE 0            // 2 layers * 32 tiles
#define W1_BASE  64           // 2 layers * 128 tiles (chunk-major: fc*32 + ntl*4 + kt)
#define W2_BASE  320          // 2 layers * 128 tiles (chunk-major: fc*32 + nt*4 + ktl)
#define NTILES_TOTAL 576
#define NSTAGES  18           // (1 GCN + 4*2 FFN chunks) * 2 layers
#define STAGE_BYTES 32768     // 32 tiles * 1KB, contiguous per stage

// ---------------------------------------------------------------------------
// Prep: convert f32 weights to f16 in WMMA B-fragment order.
// Per 32x16 (KxN) tile (contiguous-K 16-bit B order used here, wave32): lane l -> column n=l&15,
// K base = (l>=16)?16:0, half h -> K=base+h. Tile = 512 halves, lane-contig.
// ---------------------------------------------------------------------------
__global__ void prep_weights_kernel(const float* __restrict__ gcnW,
                                    const float* __restrict__ W1,
                                    const float* __restrict__ W2,
                                    _Float16* __restrict__ ws) {
  int t    = blockIdx.x * blockDim.x + threadIdx.x;
  int tile = t >> 5;
  int lid  = t & 31;
  if (tile >= NTILES_TOTAL) return;

  const float* src;
  int K, N, nt, kt;
  if (tile < W1_BASE) {                           // gcn_W: K=128 N=128
    int layer = tile / 32; int local = tile % 32;
    K = 128; N = 128; src = gcnW + layer * 128 * 128;
    nt = local >> 2; kt = local & 3;
  } else if (tile < W2_BASE) {                    // ffn_W1: K=128 N=512 (nt global 0..31)
    int i = tile - W1_BASE; int layer = i / 128; int local = i % 128;
    K = 128; N = 512; src = W1 + layer * 128 * 512;
    nt = local >> 2; kt = local & 3;
  } else {                                        // ffn_W2: K=512 N=128, chunk-major
    int i = tile - W2_BASE; int layer = i / 128; int local = i % 128;
    K = 512; N = 128; src = W2 + layer * 512 * 128;
    int fc = local >> 5; int rem = local & 31;
    nt = rem >> 2; kt = fc * 4 + (rem & 3);
  }
  int n     = nt * 16 + (lid & 15);
  int kbase = kt * 32 + ((lid & 16) ? 16 : 0);

  _Float16* dst = ws + (size_t)tile * 512 + lid * 16;
#pragma unroll
  for (int h = 0; h < 16; ++h)
    dst[h] = (_Float16)src[(size_t)(kbase + h) * N + n];
}

// ---------------------------------------------------------------------------
// Fragment helpers
// ---------------------------------------------------------------------------
__device__ inline v16h wmma_load_A(const _Float16* __restrict__ A, int r0, int k0, int lid) {
  // A 16x32 layout (contiguous K, matches B prep): lanes 0-15 -> M=lane, K{0..15};
  // lanes 16-31 -> same M, K{16..31}.
  int m     = lid & 15;
  int khalf = (lid & 16) ? 16 : 0;
  const h2* base = (const h2*)(A + (r0 + m) * LDA + k0 + khalf);
  v16h a;
  h2* ap = (h2*)&a;
#pragma unroll
  for (int j = 0; j < 8; ++j)
    ap[j] = base[j];
  return a;
}

__device__ inline v16h wmma_load_B(const _Float16* __restrict__ Wl, int tileLocal, int lid) {
  return *(const v16h*)(Wl + tileLocal * 512 + lid * 16);  // 32B contiguous per lane (LDS)
}

__device__ inline v8f wmma_mac(v16h a, v16h b, v8f c) {
  return __builtin_amdgcn_wmma_f32_16x16x32_f16(false, a, false, b, (short)0, c, false, false);
}

// ---------------------------------------------------------------------------
// TDM: DMA one contiguous 32KB weight stage (16384 f16 elements) into LDS.
// D# per CDNA5 ISA: 1-D tensor, data_size=2B.
// ---------------------------------------------------------------------------
__device__ inline void tdm_issue(const _Float16* ws, int tileBase, const _Float16* ldsDst) {
  uint64_t ga = (uint64_t)(uintptr_t)(ws + (size_t)tileBase * 512);
  uint32_t la = (uint32_t)(uintptr_t)ldsDst;     // generic LDS addr[31:0] = LDS offset

  u32x4 g0;
  g0[0] = 1u;                                    // count=1 (valid), user mode
  g0[1] = la;                                    // lds_addr
  g0[2] = (unsigned)ga;                          // global_addr[31:0]
  g0[3] = (unsigned)(ga >> 32) | 0x80000000u;    // global_addr[56:32] | type=2

  i32x8 g1;
  g1[0] = 0x00010000;                            // wg_mask=0, data_size=1 (2 bytes)
  g1[1] = (int)(16384u << 16);                   // tensor_dim0[15:0] in bits[63:48]
  g1[2] = (int)(1u << 16);                       // dim0 hi=0, tensor_dim1=1
  g1[3] = (int)(16384u << 16);                   // tile_dim0 = 16384
  g1[4] = 0;                                     // tile_dim1/2 unused
  g1[5] = 16384;                                 // tensor_dim0_stride
  g1[6] = 0; g1[7] = 0;

  i32x4 gz = {0, 0, 0, 0};
#if __clang_major__ >= 23
  i32x8 gz8 = {0, 0, 0, 0, 0, 0, 0, 0};
  __builtin_amdgcn_tensor_load_to_lds(g0, g1, gz, gz, gz8, 0);
#else
  __builtin_amdgcn_tensor_load_to_lds(g0, g1, gz, gz, 0);
#endif
}

// stage schedule: per layer l: s=0 -> GCN W; s=1,3,5,7 -> W1 chunk fc; s=2,4,6,8 -> W2 chunk fc
__device__ inline int stage_tile_base(int q) {
  int l = q / 9, s = q % 9;
  if (s == 0) return GCN_BASE + l * 32;
  int fc = (s - 1) >> 1;
  return ((s & 1) ? W1_BASE : W2_BASE) + l * 128 + fc * 32;
}

// ---------------------------------------------------------------------------
// Main fused kernel: one block = 16 graphs.
// ---------------------------------------------------------------------------
__global__ void __launch_bounds__(256, 1)
gcn_backbone_kernel(const float* __restrict__ graph,
                    const int*   __restrict__ op_idx,
                    const float* __restrict__ op_table,
                    const float* __restrict__ dev_embed,
                    const float* __restrict__ gcn_b,
                    const float* __restrict__ ln_g,
                    const float* __restrict__ ln_b,
                    const float* __restrict__ ffn_b1,
                    const float* __restrict__ ffn_b2,
                    const float* __restrict__ fc_w,
                    const float* __restrict__ fc_b,
                    const _Float16* __restrict__ ws,
                    float* __restrict__ out) {
  extern __shared__ char smem[];
  _Float16* Wb0  = (_Float16*)smem;                                // 32KB weight stage buf 0
  _Float16* Wb1  = (_Float16*)(smem + STAGE_BYTES);                // 32KB weight stage buf 1
  float*    C32  = (float*)(smem + 2 * STAGE_BYTES);               // [ROWS][LDC] f32 master / GEMM C
  float*    adjs = C32 + ROWS * LDC;                               // [ROWS][12]  norm adjacency rows
  float*    dsc  = adjs + ROWS * 12;                               // [ROWS]      rsqrt(deg)
  _Float16* A16  = (_Float16*)(dsc + ROWS);                        // [ROWS][LDA]
  _Float16* Hc16 = A16 + ROWS * LDA;                               // [ROWS][LDA] FFN hidden chunk

  const int tid = threadIdx.x;
  const int wid = tid >> 5;
  const int lid = tid & 31;
  const int g0  = blockIdx.x * GPB;

  int q = 0;  // weight stage counter (uniform)
  // kick off DMA of first weight stage; overlaps embedding + adjacency prep
  if (wid == 0) tdm_issue(ws, stage_tile_base(0), Wb0);

  // ---- 1. embedding -> C32 (x master) ----------------------------------
  for (int idx = tid; idx < ROWS * RR; idx += 256) {
    int r = idx >> 7, f = idx & 127;
    int g = g0 + r / NN, n = r % NN;
    int op = op_idx[g * NN + n];
    C32[r * LDC + f] = op_table[op * RR + f] + dev_embed[f];
  }

  // ---- 2. symmetric GCN normalization of adjacency ---------------------
  if (tid < ROWS) {
    int g = tid / NN, i = tid % NN;
    const float* gr = graph + (size_t)(g0 + g) * NN * NN + i * NN;
    float s = 0.f;
#pragma unroll
    for (int j = 0; j < NN; ++j) {
      float v = gr[j] + (i == j ? 1.0f : 0.0f);
      adjs[tid * 12 + j] = v;
      s += v;
    }
    dsc[tid] = rsqrtf(s);
  }
  __syncthreads();
  if (tid < ROWS) {
    int g = tid / NN;
    float di = dsc[tid];
#pragma unroll
    for (int j = 0; j < NN; ++j)
      adjs[tid * 12 + j] *= di * dsc[g * NN + j];
  }
  __syncthreads();

  // ---- 3. layers --------------------------------------------------------
  for (int l = 0; l < 2; ++l) {
    // 3a. message passing: A16 = (f16)(norm_adj @ x), x in C32
    for (int idx = tid; idx < ROWS * RR; idx += 256) {
      int r = idx >> 7, f = idx & 127;
      int g = r / NN;
      float s = 0.f;
#pragma unroll
      for (int j = 0; j < NN; ++j)
        s += adjs[r * 12 + j] * C32[(g * NN + j) * LDC + f];
      A16[r * LDA + f] = (_Float16)s;
    }
    __syncthreads();

    // 3b. GEMM: C32 = A16 @ gcn_W[l]   (stage q: GCN weights in LDS)
    if (wid == 0) __builtin_amdgcn_s_wait_tensorcnt(0);
    __syncthreads();
    if (wid == 0 && q + 1 < NSTAGES)
      tdm_issue(ws, stage_tile_base(q + 1), (q & 1) ? Wb0 : Wb1);  // prefetch next stage
    {
      const _Float16* Wc = (q & 1) ? Wb1 : Wb0;
      for (int t = wid; t < MT * 8; t += NWAVES) {
        int mt = t / 8, nt = t % 8;
        int r0 = mt * 16, c0 = nt * 16;
        v8f c = {};
#pragma unroll
        for (int kt = 0; kt < 4; ++kt) {
          v16h a = wmma_load_A(A16, r0, kt * 32, lid);
          v16h b = wmma_load_B(Wc, nt * 4 + kt, lid);
          c = wmma_mac(a, b, c);
        }
        int n = lid & 15, mb = (lid & 16) ? 8 : 0;
#pragma unroll
        for (int v = 0; v < 8; ++v)
          C32[(r0 + mb + v) * LDC + c0 + n] = c[v];
      }
    }
    q++;
    __syncthreads();

    // 3c. bias + relu + layernorm -> A16 (h, also the residual base)
    if (tid < ROWS) {
      const float* row = C32 + tid * LDC;
      float s = 0.f, s2 = 0.f;
      for (int f = 0; f < RR; ++f) {
        float v = row[f] + gcn_b[l * RR + f];
        v = v > 0.f ? v : 0.f;
        s += v; s2 += v * v;
      }
      float mu = s * (1.0f / RR);
      float var = s2 * (1.0f / RR) - mu * mu;
      float rs = rsqrtf(var + 1e-5f);
      for (int f = 0; f < RR; ++f) {
        float v = row[f] + gcn_b[l * RR + f];
        v = v > 0.f ? v : 0.f;
        A16[tid * LDA + f] = (_Float16)((v - mu) * rs * ln_g[l * RR + f] + ln_b[l * RR + f]);
      }
    }
    __syncthreads();

    // 3d. residual init: C32 = h + ffn_b2[l]
    for (int idx = tid; idx < ROWS * RR; idx += 256) {
      int r = idx >> 7, f = idx & 127;
      C32[r * LDC + f] = (float)A16[r * LDA + f] + ffn_b2[l * RR + f];
    }
    __syncthreads();

    // 3e. FFN in 4 chunks of 128 hidden units
    for (int fc = 0; fc < 4; ++fc) {
      // hidden chunk: Hc16 = relu(A16 @ W1[:, fc*128:+128] + b1)  (stage q, odd)
      if (wid == 0) __builtin_amdgcn_s_wait_tensorcnt(0);
      __syncthreads();
      if (wid == 0 && q + 1 < NSTAGES)
        tdm_issue(ws, stage_tile_base(q + 1), (q & 1) ? Wb0 : Wb1);
      {
        const _Float16* Wc = (q & 1) ? Wb1 : Wb0;
        for (int t = wid; t < MT * 8; t += NWAVES) {
          int mt = t / 8, ntl = t % 8;
          int r0 = mt * 16;
          v8f c = {};
#pragma unroll
          for (int kt = 0; kt < 4; ++kt) {
            v16h a = wmma_load_A(A16, r0, kt * 32, lid);
            v16h b = wmma_load_B(Wc, ntl * 4 + kt, lid);
            c = wmma_mac(a, b, c);
          }
          int n = lid & 15, mb = (lid & 16) ? 8 : 0;
          float b1v = ffn_b1[l * FF + (fc * 8 + ntl) * 16 + n];
#pragma unroll
          for (int v = 0; v < 8; ++v) {
            float x = c[v] + b1v;
            x = x > 0.f ? x : 0.f;
            Hc16[(r0 + mb + v) * LDA + ntl * 16 + n] = (_Float16)x;
          }
        }
      }
      q++;
      __syncthreads();

      // accumulate: C32 += Hc16 @ W2[fc*128:+128, :]  (stage q, even)
      if (wid == 0) __builtin_amdgcn_s_wait_tensorcnt(0);
      __syncthreads();
      if (wid == 0 && q + 1 < NSTAGES)
        tdm_issue(ws, stage_tile_base(q + 1), (q & 1) ? Wb0 : Wb1);
      {
        const _Float16* Wc = (q & 1) ? Wb1 : Wb0;
        for (int t = wid; t < MT * 8; t += NWAVES) {
          int mt = t / 8, nt = t % 8;
          int r0 = mt * 16, c0 = nt * 16;
          int n = lid & 15, mb = (lid & 16) ? 8 : 0;
          v8f c;
#pragma unroll
          for (int v = 0; v < 8; ++v)
            c[v] = C32[(r0 + mb + v) * LDC + c0 + n];
#pragma unroll
          for (int kk = 0; kk < 4; ++kk) {
            v16h a = wmma_load_A(Hc16, r0, kk * 32, lid);
            v16h b = wmma_load_B(Wc, nt * 4 + kk, lid);
            c = wmma_mac(a, b, c);
          }
#pragma unroll
          for (int v = 0; v < 8; ++v)
            C32[(r0 + mb + v) * LDC + c0 + n] = c[v];
        }
      }
      q++;
      __syncthreads();
    }
    // C32 now holds layer output x for next iteration
  }

  // ---- 4. readout: mean pool over 9 nodes, fc, sigmoid ------------------
  if (tid < GPB) {
    float acc = 0.f;
    for (int f = 0; f < RR; ++f) {
      float s = 0.f;
#pragma unroll
      for (int n = 0; n < NN; ++n)
        s += C32[(tid * NN + n) * LDC + f];
      acc += (s * (1.0f / NN)) * fc_w[f];
    }
    float z = acc + fc_b[0];
    out[g0 + tid] = 1.0f / (1.0f + __expf(-z));
  }
}

// ---------------------------------------------------------------------------
extern "C" void kernel_launch(void* const* d_in, const int* in_sizes, int n_in,
                              void* d_out, int out_size, void* d_ws, size_t ws_size,
                              hipStream_t stream) {
  const float* graph     = (const float*)d_in[0];
  const int*   op_idx    = (const int*)  d_in[1];
  const float* op_table  = (const float*)d_in[2];
  const float* dev_embed = (const float*)d_in[3];
  const float* gcn_W     = (const float*)d_in[4];
  const float* gcn_b     = (const float*)d_in[5];
  const float* ln_g      = (const float*)d_in[6];
  const float* ln_b      = (const float*)d_in[7];
  const float* ffn_W1    = (const float*)d_in[8];
  const float* ffn_b1    = (const float*)d_in[9];
  const float* ffn_W2    = (const float*)d_in[10];
  const float* ffn_b2    = (const float*)d_in[11];
  const float* fc_w      = (const float*)d_in[12];
  const float* fc_b      = (const float*)d_in[13];
  float*       out       = (float*)d_out;
  _Float16*    wsw       = (_Float16*)d_ws;   // 576 tiles * 1KB = 576 KB

  // 1) swizzle weights to WMMA B-fragment order (contiguous 32KB per GEMM stage)
  prep_weights_kernel<<<(NTILES_TOTAL * 32 + 255) / 256, 256, 0, stream>>>(
      gcn_W, ffn_W1, ffn_W2, wsw);

  // 2) fused backbone: 2048 blocks of 16 graphs
  size_t smem = 2 * (size_t)STAGE_BYTES       // TDM weight stage buffers
              + (size_t)ROWS * LDC * 4        // C32
              + (size_t)ROWS * 12 * 4         // adjacency
              + (size_t)ROWS * 4              // degree scales
              + (size_t)ROWS * LDA * 2 * 2;   // A16 + Hc16
  gcn_backbone_kernel<<<BB / GPB, 256, smem, stream>>>(
      graph, op_idx, op_table, dev_embed, gcn_b, ln_g, ln_b,
      ffn_b1, ffn_b2, fc_w, fc_b, wsw, out);
}
